// GptOssSparseMoeBlock_19980187861076
// MI455X (gfx1250) — compile-verified, matrix operand order checked by reading, not run on hardware
//
#include <hip/hip_runtime.h>
#include <hip/hip_bf16.h>
#include <math.h>

#define TOK   1024   // T = B*S
#define HID   1024   // H
#define NEXP  16     // E
#define IDIM  2880   // I
#define TOPK  4
#define TILE_T 128   // tokens per expert tile (8 WMMA M-subtiles)
#define MSUB   (TILE_T / 16)
#define NSPLIT_GU 9  // gate_up: 180 N-tiles -> 20 per partition
#define NSPLIT_DN 8  // down:     64 N-tiles -> 8 per partition (1 per wave)

typedef __attribute__((ext_vector_type(16))) __bf16       bf16x16;
typedef __attribute__((ext_vector_type(2)))  __bf16       bf16x2;
typedef __attribute__((ext_vector_type(8)))  float        f32x8;
typedef __attribute__((ext_vector_type(8)))  unsigned int u32x8;

// two f32 -> packed bf16x2 (native v_cvt_pk_bf16_f32)
static __device__ __forceinline__ unsigned pack_bf16(float a, float b) {
  bf16x2 p;
  p[0] = (__bf16)a;
  p[1] = (__bf16)b;
  return __builtin_bit_cast(unsigned, p);
}

static __device__ __forceinline__ unsigned short bf_bits(float f) {
  __bf16 h = (__bf16)f;
  return __builtin_bit_cast(unsigned short, h);
}

// A-fragment (16x32 bf16): lane holds row M=lane&15; K chunks [ka,ka+8) and [kb,kb+8)
static __device__ __forceinline__ bf16x16 pack_a(const unsigned short* rowp, int ka, int kb) {
  uint4 q0 = *(const uint4*)(rowp + ka);
  uint4 q1 = *(const uint4*)(rowp + kb);
  u32x8 u;
  u[0] = q0.x; u[1] = q0.y; u[2] = q0.z; u[3] = q0.w;
  u[4] = q1.x; u[5] = q1.y; u[6] = q1.z; u[7] = q1.w;
  return __builtin_bit_cast(bf16x16, u);
}

// B-fragment (32x16 bf16) from row-major fp32 weights W[k][n], leading dim ld.
static __device__ __forceinline__ bf16x16 load_b_f32(const float* __restrict__ W, long ld,
                                                     int ka, int kb, int col) {
  u32x8 u;
#pragma unroll
  for (int j = 0; j < 4; ++j)
    u[j] = pack_bf16(W[(long)(ka + 2 * j) * ld + col],
                     W[(long)(ka + 2 * j + 1) * ld + col]);
#pragma unroll
  for (int j = 0; j < 4; ++j)
    u[4 + j] = pack_bf16(W[(long)(kb + 2 * j) * ld + col],
                         W[(long)(kb + 2 * j + 1) * ld + col]);
  return __builtin_bit_cast(bf16x16, u);
}

#define WMMA_BF16(a, b, c) \
  __builtin_amdgcn_wmma_f32_16x16x32_bf16(false, (a), false, (b), (short)0, (c), false, false)

// ---------------------------------------------------------------------------
__global__ void zero_counts_kernel(int* counts) {
  if (threadIdx.x < NEXP) counts[threadIdx.x] = 0;
}

// ---------------------------------------------------------------------------
// Router: logits = x @ gate_w^T + gate_b ; top-4 softmax ; build expert lists.
__global__ void __launch_bounds__(128) router_kernel(
    const float* __restrict__ x, const float* __restrict__ gw, const float* __restrict__ gb,
    float* __restrict__ logits_out,
    int* __restrict__ counts, int* __restrict__ tokl, int* __restrict__ slotl,
    float* __restrict__ wgtl) {
  __shared__ float sl[8][NEXP];
  int tid = threadIdx.x;
  int e  = tid & 15;
  int lt = tid >> 4;                     // 0..7 local token
  int t  = blockIdx.x * 8 + lt;
  const float4* xp = (const float4*)(x + (long)t * HID);
  const float4* wp = (const float4*)(gw + (long)e * HID);
  float acc = gb[e];
  for (int i = 0; i < HID / 4; ++i) {
    float4 a = xp[i], b = wp[i];
    acc += a.x * b.x + a.y * b.y + a.z * b.z + a.w * b.w;
  }
  logits_out[(long)t * NEXP + e] = acc;
  sl[lt][e] = acc;
  __syncthreads();
  if (e == 0) {
    float v[NEXP];
#pragma unroll
    for (int i = 0; i < NEXP; ++i) v[i] = sl[lt][i];
    float tv[TOPK]; int ti[TOPK]; unsigned used = 0;
#pragma unroll
    for (int k = 0; k < TOPK; ++k) {     // argmax with lowest-index tie-break
      float best = -INFINITY; int bi = 0;
      for (int i = 0; i < NEXP; ++i)
        if (!((used >> i) & 1u) && v[i] > best) { best = v[i]; bi = i; }
      used |= 1u << bi; tv[k] = best; ti[k] = bi;
    }
    float mx = tv[0], s = 0.f, p[TOPK];
#pragma unroll
    for (int k = 0; k < TOPK; ++k) { p[k] = __expf(tv[k] - mx); s += p[k]; }
    float inv = 1.f / s;
    for (int k = 0; k < TOPK; ++k) {
      int ex = ti[k];
      int pos = atomicAdd(&counts[ex], 1);
      tokl[ex * TOK + pos]  = t;
      slotl[ex * TOK + pos] = k;
      wgtl[ex * TOK + pos]  = p[k] * inv;
    }
  }
}

// ---------------------------------------------------------------------------
// gate_up GEMM + SiLU*up, per (expert, 128-token tile, N-partition).
__global__ void __launch_bounds__(256) gateup_kernel(
    const float* __restrict__ x, const float* __restrict__ gup, const float* __restrict__ gub,
    const int* __restrict__ counts, const int* __restrict__ tokl, const int* __restrict__ slotl,
    unsigned short* __restrict__ inter) {
  int e = blockIdx.y;
  int cnt = counts[e];
  int tile0 = blockIdx.x * TILE_T;
  if (tile0 >= cnt) return;

  __shared__ __align__(16) unsigned short lx[TILE_T * HID];  // 256 KB bf16 X tile
  __shared__ int s_tok[TILE_T];
  __shared__ int s_slot[TILE_T];

  int tid = threadIdx.x;
  if (tid < TILE_T) {
    int r = tile0 + tid;
    if (r < cnt) { s_tok[tid] = tokl[e * TOK + r]; s_slot[tid] = slotl[e * TOK + r]; }
    else         { s_tok[tid] = -1;                s_slot[tid] = 0; }
  }
  __syncthreads();
  // stage X tile (fp32 -> bf16) into LDS
  for (int i = tid; i < TILE_T * (HID / 4); i += 256) {
    int row = i >> 8;              // HID/4 == 256
    int c   = (i & 255) << 2;
    int tk  = s_tok[row];
    unsigned lo = 0u, hi = 0u;
    if (tk >= 0) {
      float4 v = *(const float4*)(x + (long)tk * HID + c);
      lo = pack_bf16(v.x, v.y);
      hi = pack_bf16(v.z, v.w);
    }
    *(uint2*)(&lx[row * HID + c]) = make_uint2(lo, hi);
  }
  __syncthreads();

  int lane = tid & 31, wave = tid >> 5;
  int lg = lane >> 4, ln = lane & 15;
  const float* W = gup + (long)e * HID * (2 * IDIM);
  const long   ld = 2 * IDIM;
  const f32x8 vzero = {};
  const int per = (IDIM / 16) / NSPLIT_GU;     // 20 N-tiles per partition
  const int nt0 = blockIdx.z * per;

  for (int ntl = wave; ntl < per; ntl += 8) {
    int nt = nt0 + ntl;
    int fg = nt * 16 + ln;           // gate column
    int fu = IDIM + nt * 16 + ln;    // up column
    float bg = gub[(long)e * 2 * IDIM + fg];
    float bu = gub[(long)e * 2 * IDIM + fu];
    f32x8 cg[MSUB], cu[MSUB];
#pragma unroll
    for (int m = 0; m < MSUB; ++m) { cg[m] = vzero; cu[m] = vzero; }

    for (int k = 0; k < HID; k += 32) {
      int ka = k + lg * 8, kb = k + 16 + lg * 8;
      __builtin_prefetch(W + (long)(k + 32) * ld + fg, 0, 1);  // speculative, non-faulting
      bf16x16 bgf = load_b_f32(W, ld, ka, kb, fg);
      bf16x16 buf = load_b_f32(W, ld, ka, kb, fu);
#pragma unroll
      for (int m = 0; m < MSUB; ++m) {
        bf16x16 a = pack_a(&lx[(m * 16 + ln) * HID], ka, kb);
        cg[m] = WMMA_BF16(a, bgf, cg[m]);
        cu[m] = WMMA_BF16(a, buf, cu[m]);
      }
    }
#pragma unroll
    for (int m = 0; m < MSUB; ++m) {
#pragma unroll
      for (int r = 0; r < 8; ++r) {
        int row = m * 16 + r + 8 * lg;   // D layout: M = r + 8*(lane>=16)
        int tk = s_tok[row];
        if (tk >= 0) {
          float g = cg[m][r] + bg;
          float u = cu[m][r] + bu;
          float val = (g / (1.f + __expf(-g))) * u;   // silu(g) * u
          inter[(long)(tk * TOPK + s_slot[row]) * IDIM + nt * 16 + ln] = bf_bits(val);
        }
      }
    }
  }
}

// ---------------------------------------------------------------------------
// down GEMM: partial[(t*4+slot)*H + h] = w * (inter @ down_proj + bias)
__global__ void __launch_bounds__(256) down_kernel(
    const float* __restrict__ dw, const float* __restrict__ db,
    const int* __restrict__ counts, const int* __restrict__ tokl, const int* __restrict__ slotl,
    const float* __restrict__ wgtl, const unsigned short* __restrict__ inter,
    float* __restrict__ partial) {
  int e = blockIdx.y;
  int cnt = counts[e];
  int tile0 = blockIdx.x * TILE_T;
  if (tile0 >= cnt) return;

  __shared__ int   s_tok[TILE_T];
  __shared__ int   s_slot[TILE_T];
  __shared__ float s_w[TILE_T];
  int tid = threadIdx.x;
  if (tid < TILE_T) {
    int r = tile0 + tid;
    if (r < cnt) { s_tok[tid] = tokl[e*TOK+r]; s_slot[tid] = slotl[e*TOK+r]; s_w[tid] = wgtl[e*TOK+r]; }
    else         { s_tok[tid] = -1; s_slot[tid] = 0; s_w[tid] = 0.f; }
  }
  __syncthreads();

  int lane = tid & 31, wave = tid >> 5;
  int lg = lane >> 4, ln = lane & 15;
  const float* W = dw + (long)e * IDIM * HID;
  const f32x8 vzero = {};
  const u32x8 uzero = {};
  const int per = (HID / 16) / NSPLIT_DN;      // 8 N-tiles per partition
  const int nt0 = blockIdx.z * per;

  const unsigned short* abase[MSUB];
#pragma unroll
  for (int m = 0; m < MSUB; ++m) {
    int tk = s_tok[m * 16 + ln];
    abase[m] = (tk >= 0) ? inter + (long)(tk * TOPK + s_slot[m * 16 + ln]) * IDIM : nullptr;
  }

  for (int ntl = wave; ntl < per; ntl += 8) {
    int nt = nt0 + ntl;
    int h = nt * 16 + ln;
    float bias = db[(long)e * HID + h];
    f32x8 c[MSUB];
#pragma unroll
    for (int m = 0; m < MSUB; ++m) c[m] = vzero;

    for (int k = 0; k < IDIM; k += 32) {
      int ka = k + lg * 8, kb = k + 16 + lg * 8;
      __builtin_prefetch(W + (long)(k + 32) * HID + h, 0, 1);  // speculative
      bf16x16 b = load_b_f32(W, HID, ka, kb, h);
#pragma unroll
      for (int m = 0; m < MSUB; ++m) {
        bf16x16 a;
        if (abase[m]) a = pack_a(abase[m], ka, kb);
        else          a = __builtin_bit_cast(bf16x16, uzero);
        c[m] = WMMA_BF16(a, b, c[m]);
      }
    }
#pragma unroll
    for (int m = 0; m < MSUB; ++m) {
#pragma unroll
      for (int r = 0; r < 8; ++r) {
        int row = m * 16 + r + 8 * lg;
        int tk = s_tok[row];
        if (tk >= 0)
          partial[(long)(tk * TOPK + s_slot[row]) * HID + h] = s_w[row] * (c[m][r] + bias);
      }
    }
  }
}

// ---------------------------------------------------------------------------
// out[t,h] = sum_{k=0..3} partial[t,k,h]  (fixed order -> deterministic)
__global__ void __launch_bounds__(256) combine_kernel(const float* __restrict__ partial,
                                                      float* __restrict__ out) {
  long i = (long)blockIdx.x * blockDim.x + threadIdx.x;  // one float4
  long o = i * 4;
  int t = (int)(o / HID);
  int h = (int)(o % HID);
  const float* base = partial + (long)t * TOPK * HID + h;
  float4 a = *(const float4*)(base + 0 * HID);
  float4 b = *(const float4*)(base + 1 * HID);
  float4 c = *(const float4*)(base + 2 * HID);
  float4 d = *(const float4*)(base + 3 * HID);
  float4 r;
  r.x = ((a.x + b.x) + c.x) + d.x;
  r.y = ((a.y + b.y) + c.y) + d.y;
  r.z = ((a.z + b.z) + c.z) + d.z;
  r.w = ((a.w + b.w) + c.w) + d.w;
  *(float4*)(out + o) = r;
}

// ---------------------------------------------------------------------------
extern "C" void kernel_launch(void* const* d_in, const int* in_sizes, int n_in,
                              void* d_out, int out_size, void* d_ws, size_t ws_size,
                              hipStream_t stream) {
  (void)in_sizes; (void)n_in; (void)out_size; (void)ws_size;
  const float* x   = (const float*)d_in[0];   // hidden_states (1,1024,1024)
  const float* gw  = (const float*)d_in[1];   // gate_w (E,H)
  const float* gb  = (const float*)d_in[2];   // gate_b (E)
  const float* gup = (const float*)d_in[3];   // gate_up_proj (E,H,2I)
  const float* gub = (const float*)d_in[4];   // gate_up_proj_bias (E,2I)
  const float* dw  = (const float*)d_in[5];   // down_proj (E,I,H)
  const float* db  = (const float*)d_in[6];   // down_proj_bias (E,H)

  float* out    = (float*)d_out;              // (1,S,H) fp32
  float* logits = out + (long)TOK * HID;      // (T,E) fp32, concatenated

  // workspace layout (~40.6 MB)
  int*   counts = (int*)d_ws;                          // 64 ints (16 used, padded)
  int*   tokl   = counts + 64;                         // E*T
  int*   slotl  = tokl + NEXP * TOK;                   // E*T
  float* wgtl   = (float*)(slotl + NEXP * TOK);        // E*T
  unsigned short* inter = (unsigned short*)(wgtl + NEXP * TOK);     // T*4*I bf16
  float* partial = (float*)(inter + (long)TOK * TOPK * IDIM);       // T*4*H f32

  zero_counts_kernel<<<dim3(1), dim3(64), 0, stream>>>(counts);
  router_kernel<<<dim3(TOK / 8), dim3(128), 0, stream>>>(x, gw, gb, logits,
                                                         counts, tokl, slotl, wgtl);
  gateup_kernel<<<dim3(TOK / TILE_T, NEXP, NSPLIT_GU), dim3(256), 0, stream>>>(
      x, gup, gub, counts, tokl, slotl, inter);
  down_kernel<<<dim3(TOK / TILE_T, NEXP, NSPLIT_DN), dim3(256), 0, stream>>>(
      dw, db, counts, tokl, slotl, wgtl, inter, partial);
  combine_kernel<<<dim3((TOK * HID / 4) / 256), dim3(256), 0, stream>>>(partial, out);
}
